// WSchnet_N_13443247637170
// MI455X (gfx1250) — matrix operand order repaired and checked by reading, not compiled, not run on hardware
//
#include <hip/hip_runtime.h>

// ---------------------------------------------------------------------------
// WSchnet on MI455X (gfx1250, wave32, WMMA).
// All 64x64 GEMMs run through v_wmma_f32_16x16x32_f16 (f16 in, f32 acc).
// Edge stage (RBF -> filter MLP -> gather -> mul -> scatter) is fully fused:
// h [E,64] is never materialized. node/new_node/agg (~38MB) are L2-resident.
// ---------------------------------------------------------------------------

typedef __attribute__((ext_vector_type(16))) _Float16 v16h;
typedef __attribute__((ext_vector_type(8)))  float    v8f;

#define DIM   64
#define WPB   8          // waves per block (256 threads)
#define LOG2F_CONST 0.6931471805599453f

__device__ __forceinline__ float softplus_f(float x) {
  // stable log(1+exp(x))
  return fmaxf(x, 0.0f) + log1pf(__expf(-fabsf(x)));
}
__device__ __forceinline__ float sp05_f(float x) {   // Softplus(beta=0.5)
  return 2.0f * softplus_f(0.5f * x);
}

__device__ __forceinline__ v8f wmma16x16x32(v16h a, v16h b, v8f c) {
  return __builtin_amdgcn_wmma_f32_16x16x32_f16(
      /*neg_a=*/false, a, /*neg_b=*/false, b,
      /*c_mod=*/(short)0, c, /*reuse_a=*/false, /*reuse_b=*/false);
}

// A-operand (16x32 f16) from an f32 row-major [rows, 64] matrix.
// lane: r = lane&15 (row), kb = (lane>>4)*8.
// elements 0..7  -> K = kc + kb + e
// elements 8..15 -> K = kc + 16 + kb + e
__device__ __forceinline__ v16h load_A_f32(const float* __restrict__ m,
                                           int row, int kc, int kb) {
  const float* p0 = m + (size_t)row * DIM + kc + kb;
  v16h a;
#pragma unroll
  for (int e = 0; e < 8; ++e) {
    a[e]     = (_Float16)p0[e];
    a[8 + e] = (_Float16)p0[16 + e];
  }
  return a;
}

// B-operand fetch from pre-packed weights: [kchunk][ntile][lane][16 halves]
__device__ __forceinline__ v16h load_B(const _Float16* __restrict__ wp,
                                       int kc, int t, int lane, int ntiles) {
  return *(const v16h*)(wp + ((size_t)(kc * ntiles + t) * 32 + lane) * 16);
}

// ---------------------------------------------------------------------------
// Pack f32 weight [64, ncols] (row-major, K-major) into WMMA B-operand layout:
// dst[((kc*ntiles + t)*32 + lane)*16 + e] = (f16) w[kc*32 + (lane>>4)*8 + e + (e>=8 ? 8 : 0)]
//                                                 [t*16 + (lane&15)]
// Zero-pads columns >= ncols (used for w_cls, 100 -> 112).
// ---------------------------------------------------------------------------
__global__ void pack_weight_kernel(const float* __restrict__ w,
                                   _Float16* __restrict__ dst,
                                   int ncols, int ntiles) {
  int tid = blockIdx.x * blockDim.x + threadIdx.x;
  int total = 2 * ntiles * 32;
  if (tid >= total) return;
  int lane = tid & 31;
  int grp  = tid >> 5;
  int t    = grp % ntiles;
  int kc   = grp / ntiles;
  int n    = t * 16 + (lane & 15);
  int kb   = kc * 32 + (lane >> 4) * 8;
  v16h v;
#pragma unroll
  for (int e = 0; e < 16; ++e) {
    float x = (n < ncols) ? w[(size_t)(kb + e + ((e >= 8) ? 8 : 0)) * ncols + n] : 0.0f;
    v[e] = (_Float16)x;
  }
  *(v16h*)(dst + (size_t)tid * 16) = v;
}

// node[i] = emb[node_type[i]]                                   [N,64] f32
__global__ void embed_kernel(const float* __restrict__ emb,
                             const int* __restrict__ node_type,
                             float* __restrict__ node, int total) {
  int i = blockIdx.x * blockDim.x + threadIdx.x;
  if (i < total) {
    int n = i >> 6, c = i & 63;
    node[i] = emb[(size_t)node_type[n] * DIM + c];
  }
}

__global__ void zero_kernel(float* __restrict__ p, int total) {
  int i = blockIdx.x * blockDim.x + threadIdx.x;
  if (i < total) p[i] = 0.0f;
}

// ---------------------------------------------------------------------------
// new_node = node @ w1   (no bias).  One wave per 16-row tile, 8 WMMAs.
// ---------------------------------------------------------------------------
__global__ void __launch_bounds__(256)
node1_gemm_kernel(const float* __restrict__ node,
                  const _Float16* __restrict__ w1p,
                  float* __restrict__ out, int N) {
  int wave = (int)((blockIdx.x * blockDim.x + threadIdx.x) >> 5);
  int lane = threadIdx.x & 31;
  int row0 = wave * 16;
  if (row0 >= N) return;
  int r = lane & 15, hi = lane >> 4, kb = hi * 8;
  int arow = row0 + r; if (arow >= N) arow = N - 1;

  v8f acc[4] = {};
#pragma unroll
  for (int kc = 0; kc < 2; ++kc) {
    v16h a = load_A_f32(node, arow, kc * 32, kb);
#pragma unroll
    for (int t = 0; t < 4; ++t)
      acc[t] = wmma16x16x32(a, load_B(w1p, kc, t, lane, 4), acc[t]);
  }
#pragma unroll
  for (int t = 0; t < 4; ++t) {
    int col = t * 16 + r;
#pragma unroll
    for (int rr = 0; rr < 8; ++rr) {
      int m = row0 + rr + hi * 8;
      if (m < N) out[(size_t)m * DIM + col] = acc[t][rr];
    }
  }
}

// ---------------------------------------------------------------------------
// Fused edge stage, one wave per 16 edges:
//   rbf[k]  = exp(-(d - 1.25k)^2 / 1.25)                     (5 centers)
//   tA      = sp05(rbf @ cf_w1 + cf_b1)   built directly in A-operand layout
//   h       = tA @ cf_w2 + cf_b2          (WMMA, 8x)
//   atomicAdd(agg[dst], new_node[src] * h)
// ---------------------------------------------------------------------------
__global__ void __launch_bounds__(256)
edge_kernel(const float* __restrict__ distance,
            const float* __restrict__ cf_w1,   // [5,64]
            const float* __restrict__ cf_b1,   // [64]
            const _Float16* __restrict__ w2p,  // packed [2][4][32][16]
            const float* __restrict__ cf_b2,   // [64]
            const float* __restrict__ newn,    // [N,64]
            const int* __restrict__ src,
            const int* __restrict__ dstv,
            float* __restrict__ agg, int E) {
  int wave = (int)((blockIdx.x * blockDim.x + threadIdx.x) >> 5);
  int lane = threadIdx.x & 31;
  int e0 = wave * 16;
  if (e0 >= E) return;
  int r = lane & 15, hi = lane >> 4, kb = hi * 8;

  int er = e0 + r; if (er >= E) er = E - 1;
  float d = distance[er];
  float rbf[5];
#pragma unroll
  for (int k = 0; k < 5; ++k) {
    float dd = d - 1.25f * (float)k;
    rbf[k] = __expf(-dd * dd * 0.8f);           // gap = 1.25
  }

  v8f acc[4] = {};
#pragma unroll
  for (int kc = 0; kc < 2; ++kc) {
    v16h a;
#pragma unroll
    for (int e = 0; e < 16; ++e) {
      int j = kc * 32 + ((e < 8) ? 0 : 16) + kb + (e & 7);
      float x = cf_b1[j];
#pragma unroll
      for (int k = 0; k < 5; ++k) x = fmaf(rbf[k], cf_w1[k * DIM + j], x);
      a[e] = (_Float16)sp05_f(x);
    }
#pragma unroll
    for (int t = 0; t < 4; ++t)
      acc[t] = wmma16x16x32(a, load_B(w2p, kc, t, lane, 4), acc[t]);
  }

  int se[8], de[8];
#pragma unroll
  for (int rr = 0; rr < 8; ++rr) {
    int m = e0 + rr + hi * 8; if (m >= E) m = E - 1;
    se[rr] = src[m];
    de[rr] = dstv[m];
  }
#pragma unroll
  for (int t = 0; t < 4; ++t) {
    int col = t * 16 + r;
    float b2 = cf_b2[col];
#pragma unroll
    for (int rr = 0; rr < 8; ++rr) {
      int m = e0 + rr + hi * 8;
      if (m < E) {
        float h = acc[t][rr] + b2;
        float val = h * newn[(size_t)se[rr] * DIM + col];
        atomicAdd(&agg[(size_t)de[rr] * DIM + col], val);
      }
    }
  }
}

// ---------------------------------------------------------------------------
// cf = sp05(agg @ w2 + b2) ; node += cf @ w3 + b3
// Two WMMA stages chained through a per-wave LDS tile (D-layout -> A-layout).
// ---------------------------------------------------------------------------
__global__ void __launch_bounds__(256)
node23_kernel(const float* __restrict__ agg,
              const _Float16* __restrict__ w2p, const float* __restrict__ b2,
              const _Float16* __restrict__ w3p, const float* __restrict__ b3,
              float* __restrict__ node, int N) {
  __shared__ _Float16 lds[WPB][16 * DIM];
  int wv   = threadIdx.x >> 5;
  int lane = threadIdx.x & 31;
  int wave = blockIdx.x * WPB + wv;
  int row0 = wave * 16;
  int r = lane & 15, hi = lane >> 4, kb = hi * 8;
  bool active = (row0 < N);

  if (active) {
    int arow = row0 + r; if (arow >= N) arow = N - 1;
    v8f acc[4] = {};
#pragma unroll
    for (int kc = 0; kc < 2; ++kc) {
      v16h a = load_A_f32(agg, arow, kc * 32, kb);
#pragma unroll
      for (int t = 0; t < 4; ++t)
        acc[t] = wmma16x16x32(a, load_B(w2p, kc, t, lane, 4), acc[t]);
    }
#pragma unroll
    for (int t = 0; t < 4; ++t) {
      int col = t * 16 + r;
      float bb = b2[col];
#pragma unroll
      for (int rr = 0; rr < 8; ++rr) {
        int m = rr + hi * 8;                       // local row in tile
        lds[wv][m * DIM + col] = (_Float16)sp05_f(acc[t][rr] + bb);
      }
    }
  }
  __syncthreads();
  if (active) {
    v8f acc[4] = {};
#pragma unroll
    for (int kc = 0; kc < 2; ++kc) {
      v16h a;
#pragma unroll
      for (int e = 0; e < 8; ++e) {
        a[e]     = lds[wv][r * DIM + kc * 32 + kb + e];
        a[8 + e] = lds[wv][r * DIM + kc * 32 + 16 + kb + e];
      }
#pragma unroll
      for (int t = 0; t < 4; ++t)
        acc[t] = wmma16x16x32(a, load_B(w3p, kc, t, lane, 4), acc[t]);
    }
#pragma unroll
    for (int t = 0; t < 4; ++t) {
      int col = t * 16 + r;
      float bb = b3[col];
#pragma unroll
      for (int rr = 0; rr < 8; ++rr) {
        int m = row0 + rr + hi * 8;
        if (m < N) node[(size_t)m * DIM + col] += acc[t][rr] + bb;   // residual
      }
    }
  }
}

// ---------------------------------------------------------------------------
// atom = node @ w_d1 + b_d1 ; r = softplus(atom) - log2 ; out = r @ w_cls + b
// w_cls packed as 7 N-tiles (100 -> 112, zero-padded); stores masked col<100.
// ---------------------------------------------------------------------------
__global__ void __launch_bounds__(256)
head_kernel(const float* __restrict__ node,
            const _Float16* __restrict__ wd1p, const float* __restrict__ bd1,
            const _Float16* __restrict__ wclsp, const float* __restrict__ bcls,
            float* __restrict__ out, int N) {
  __shared__ _Float16 lds[WPB][16 * DIM];
  int wv   = threadIdx.x >> 5;
  int lane = threadIdx.x & 31;
  int wave = blockIdx.x * WPB + wv;
  int row0 = wave * 16;
  int r = lane & 15, hi = lane >> 4, kb = hi * 8;
  bool active = (row0 < N);

  if (active) {
    int arow = row0 + r; if (arow >= N) arow = N - 1;
    v8f acc[4] = {};
#pragma unroll
    for (int kc = 0; kc < 2; ++kc) {
      v16h a = load_A_f32(node, arow, kc * 32, kb);
#pragma unroll
      for (int t = 0; t < 4; ++t)
        acc[t] = wmma16x16x32(a, load_B(wd1p, kc, t, lane, 4), acc[t]);
    }
#pragma unroll
    for (int t = 0; t < 4; ++t) {
      int col = t * 16 + r;
      float bb = bd1[col];
#pragma unroll
      for (int rr = 0; rr < 8; ++rr) {
        int m = rr + hi * 8;
        float x = softplus_f(acc[t][rr] + bb) - LOG2F_CONST;  // ShiftSoftplus
        lds[wv][m * DIM + col] = (_Float16)x;
      }
    }
  }
  __syncthreads();
  if (active) {
    v16h a0, a1;
#pragma unroll
    for (int e = 0; e < 8; ++e) {
      a0[e]     = lds[wv][r * DIM + kb + e];
      a0[8 + e] = lds[wv][r * DIM + 16 + kb + e];
      a1[e]     = lds[wv][r * DIM + 32 + kb + e];
      a1[8 + e] = lds[wv][r * DIM + 48 + kb + e];
    }
#pragma unroll
    for (int t = 0; t < 7; ++t) {
      v8f acc = {};
      acc = wmma16x16x32(a0, load_B(wclsp, 0, t, lane, 7), acc);
      acc = wmma16x16x32(a1, load_B(wclsp, 1, t, lane, 7), acc);
      int col = t * 16 + r;
      if (col < 100) {
        float bb = bcls[col];
#pragma unroll
        for (int rr = 0; rr < 8; ++rr) {
          int m = row0 + rr + hi * 8;
          if (m < N) out[(size_t)m * 100 + col] = acc[rr] + bb;
        }
      }
    }
  }
}

// ---------------------------------------------------------------------------
extern "C" void kernel_launch(void* const* d_in, const int* in_sizes, int n_in,
                              void* d_out, int out_size, void* d_ws, size_t ws_size,
                              hipStream_t stream) {
  const float* distance = (const float*)d_in[0];
  const float* emb      = (const float*)d_in[1];
  const float* w_node1  = (const float*)d_in[2];   // [3,64,64]
  const float* cf_w1    = (const float*)d_in[3];   // [3,5,64]
  const float* cf_b1    = (const float*)d_in[4];   // [3,64]
  const float* cf_w2    = (const float*)d_in[5];   // [3,64,64]
  const float* cf_b2    = (const float*)d_in[6];   // [3,64]
  const float* w_node2  = (const float*)d_in[7];   // [3,64,64]
  const float* b_node2  = (const float*)d_in[8];   // [3,64]
  const float* w_node3  = (const float*)d_in[9];   // [3,64,64]
  const float* b_node3  = (const float*)d_in[10];  // [3,64]
  const float* w_d1     = (const float*)d_in[11];  // [64,64]
  const float* b_d1     = (const float*)d_in[12];  // [64]
  const float* w_cls    = (const float*)d_in[13];  // [64,100]
  const float* b_cls    = (const float*)d_in[14];  // [100]
  const int*   srcp     = (const int*)d_in[15];
  const int*   dstp     = (const int*)d_in[16];
  const int*   node_type= (const int*)d_in[17];

  const int E = in_sizes[0];       // distance is [E,1]
  const int N = in_sizes[17];      // node_type is [N]
  float* out = (float*)d_out;

  // Workspace carve-up (256B aligned slices).
  char* ws = (char*)d_ws;
  auto carve = [&](size_t bytes) {
    char* p = ws;
    ws += (bytes + 255) & ~(size_t)255;
    return p;
  };
  float* node = (float*)carve((size_t)N * DIM * sizeof(float));
  float* newn = (float*)carve((size_t)N * DIM * sizeof(float));
  float* aggp = (float*)carve((size_t)N * DIM * sizeof(float));
  _Float16* packs = (_Float16*)carve((size_t)13 * 4096 * sizeof(_Float16));
  _Float16* wclsp = (_Float16*)carve((size_t)2 * 7 * 32 * 16 * sizeof(_Float16));
  (void)ws_size;

  // Pack weights into WMMA B-operand layout (tiny one-shot kernels).
  for (int i = 0; i < 3; ++i) {
    pack_weight_kernel<<<1, 256, 0, stream>>>(w_node1 + (size_t)i * 4096, packs + (size_t)(0 + i) * 4096, 64, 4);
    pack_weight_kernel<<<1, 256, 0, stream>>>(cf_w2   + (size_t)i * 4096, packs + (size_t)(3 + i) * 4096, 64, 4);
    pack_weight_kernel<<<1, 256, 0, stream>>>(w_node2 + (size_t)i * 4096, packs + (size_t)(6 + i) * 4096, 64, 4);
    pack_weight_kernel<<<1, 256, 0, stream>>>(w_node3 + (size_t)i * 4096, packs + (size_t)(9 + i) * 4096, 64, 4);
  }
  pack_weight_kernel<<<1, 256, 0, stream>>>(w_d1, packs + (size_t)12 * 4096, 64, 4);
  pack_weight_kernel<<<2, 256, 0, stream>>>(w_cls, wclsp, 100, 7);   // 448 threads

  const int total = N * DIM;
  embed_kernel<<<(total + 255) / 256, 256, 0, stream>>>(emb, node_type, node, total);

  const int row_tiles  = (N + 15) / 16;                 // one wave per tile
  const int gemm_blocks = (row_tiles + WPB - 1) / WPB;
  const int edge_tiles = (E + 15) / 16;
  const int edge_blocks = (edge_tiles + WPB - 1) / WPB;

  for (int i = 0; i < 3; ++i) {
    zero_kernel<<<(total + 255) / 256, 256, 0, stream>>>(aggp, total);
    node1_gemm_kernel<<<gemm_blocks, 256, 0, stream>>>(
        node, packs + (size_t)(0 + i) * 4096, newn, N);
    edge_kernel<<<edge_blocks, 256, 0, stream>>>(
        distance, cf_w1 + (size_t)i * 5 * DIM, cf_b1 + (size_t)i * DIM,
        packs + (size_t)(3 + i) * 4096, cf_b2 + (size_t)i * DIM,
        newn, srcp, dstp, aggp, E);
    node23_kernel<<<gemm_blocks, 256, 0, stream>>>(
        aggp, packs + (size_t)(6 + i) * 4096, b_node2 + (size_t)i * DIM,
        packs + (size_t)(9 + i) * 4096, b_node3 + (size_t)i * DIM, node, N);
  }

  head_kernel<<<gemm_blocks, 256, 0, stream>>>(
      node, packs + (size_t)12 * 4096, b_d1, wclsp, b_cls, out, N);
}
